// deformable_transformer_layer_88776974008705
// MI455X (gfx1250) — matrix operand order repaired and checked by reading, not compiled, not run on hardware
//
#include <hip/hip_runtime.h>
#include <math.h>

// ---------------------------------------------------------------------------
// Types for CDNA5 WMMA (wave32, 16x16x32 bf16 -> f32)
// ---------------------------------------------------------------------------
typedef __bf16          v16bf __attribute__((ext_vector_type(16)));
typedef unsigned short  v16us __attribute__((ext_vector_type(16)));
typedef unsigned short  v8us  __attribute__((ext_vector_type(8)));
typedef float           v8f   __attribute__((ext_vector_type(8)));

static __device__ __forceinline__ unsigned short f2bf(float f) {
  unsigned int u = __float_as_uint(f);
  u += 0x7fffu + ((u >> 16) & 1u);          // round-to-nearest-even
  return (unsigned short)(u >> 16);
}

// A fragment K index: contiguous (lanes 0-15: K=0..15, lanes 16-31: K=16..31)
static __device__ __forceinline__ int kidx_a(int v, int half) {
  return half * 16 + v * 2;
}
// B fragment K index: 16-bit B 32x16 (lanes 0-15: K=0..15, lanes 16-31: K=16..31)
static __device__ __forceinline__ int kidx_b(int v, int half) {
  return half * 16 + v * 2;
}

static __device__ __forceinline__ v8f wmma_bf16(v16us a, v16us b, v8f c) {
  return __builtin_amdgcn_wmma_f32_16x16x32_bf16(
      false, __builtin_bit_cast(v16bf, a),
      false, __builtin_bit_cast(v16bf, b),
      (short)0, c, false, false);
}

static __device__ __forceinline__ float gelu_exact(float v) {
  return 0.5f * v * (1.0f + erff(v * 0.70710678118654752f));
}

// ---------------------------------------------------------------------------
// Generic batched WMMA GEMM:  C[M,N] = act(A[M,K] * B[K,N] + bias) + R
//
// Requirements (met by every call below): a_cs == 1 (A is K-contiguous and
// each lane's 8-element runs are 16B aligned), N % 64 == 0, K % 32 == 0,
// M % 16 == 0, N/64 and batch inner-counts are powers of two. When b_rs == 1
// (K-contiguous B, i.e. weight matrices stored [n][k]) b_cs must be a
// multiple of 8 so fragments load as two global_load_b128 per lane.
//
// Each wave owns a 16x64 strip of C: one A fragment is reused by four
// back-to-back v_wmma ops per 32-wide k-step.
// Batch offset: off = (batch >> bsh) * bo + (batch & bmk) * bi.
// ---------------------------------------------------------------------------
struct GArgs {
  const unsigned short* A; long long a_rs, a_bo, a_bi; int a_bsh, a_bmk;
  const unsigned short* B; long long b_rs, b_cs, b_bo, b_bi; int b_bsh, b_bmk;
  float* C;                long long c_rs, c_cs, c_bo, c_bi; int c_bsh, c_bmk;
  unsigned short* Cbf;     long long cb_rs, cb_cs, cb_bo, cb_bi; int cb_bsh, cb_bmk;
  const float* bias;
  const float* R;          long long r_rs, r_cs, r_bo, r_bi; int r_bsh, r_bmk;
  int act;                 // 0=none, 1=relu, 2=gelu(exact)
  float bf_scale;
  int K, ns_shift, ns_mask, tiles_per_batch;
};

__global__ __launch_bounds__(256)
void wmma_gemm_kernel(GArgs g) {
  const int lane = threadIdx.x & 31;
  const int wave = threadIdx.x >> 5;
  const int half = lane >> 4;
  const int mn   = lane & 15;
  const int batch = blockIdx.x;
  const int tile  = blockIdx.y * 8 + wave;
  if (tile >= g.tiles_per_batch) return;
  const int tm = tile >> g.ns_shift;
  const int n0 = (tile & g.ns_mask) * 64;
  const int m0 = tm * 16;

  const unsigned short* Ab = g.A + (long long)(batch >> g.a_bsh) * g.a_bo +
                                   (long long)(batch & g.a_bmk) * g.a_bi;
  const unsigned short* Bb = g.B + (long long)(batch >> g.b_bsh) * g.b_bo +
                                   (long long)(batch & g.b_bmk) * g.b_bi;
  const long long arow = (long long)(m0 + mn) * g.a_rs + half * 16;
  const bool b_fast = (g.b_rs == 1);   // wave-uniform

  v8f acc[4] = {{}, {}, {}, {}};
  for (int k0 = 0; k0 < g.K; k0 += 32) {
    // A fragment: two contiguous 16B runs per lane
    const v8us alo = *(const v8us*)(Ab + arow + k0);
    const v8us ahi = *(const v8us*)(Ab + arow + k0 + 8);
    v16us au;
#pragma unroll
    for (int e = 0; e < 8; ++e) { au[e] = alo[e]; au[8 + e] = ahi[e]; }
    if (k0 + 32 < g.K) {
      __builtin_prefetch(Ab + arow + k0 + 32, 0, 1);
      __builtin_prefetch(Bb + (long long)(k0 + 32 + half * 16) * g.b_rs +
                             (long long)(n0 + mn) * g.b_cs, 0, 1);
    }
    if (b_fast) {
      // K-contiguous B: per-lane 32B run -> two global_load_b128
#pragma unroll
      for (int t = 0; t < 4; ++t) {
        const unsigned short* bp = Bb + (long long)(n0 + t * 16 + mn) * g.b_cs +
                                   k0 + half * 16;
        const v8us blo = *(const v8us*)(bp);
        const v8us bhi = *(const v8us*)(bp + 8);
        v16us bu;
#pragma unroll
        for (int e = 0; e < 8; ++e) { bu[e] = blo[e]; bu[8 + e] = bhi[e]; }
        acc[t] = wmma_bf16(au, bu, acc[t]);
      }
    } else {
      // K-major B: lane-coalesced u16 gathers
#pragma unroll
      for (int t = 0; t < 4; ++t) {
        const long long bcol = (long long)(n0 + t * 16 + mn) * g.b_cs;
        v16us bu;
#pragma unroll
        for (int v = 0; v < 8; ++v) {
          const int kb = k0 + kidx_b(v, half);
          bu[2 * v]     = Bb[(long long)(kb)     * g.b_rs + bcol];
          bu[2 * v + 1] = Bb[(long long)(kb + 1) * g.b_rs + bcol];
        }
        acc[t] = wmma_bf16(au, bu, acc[t]);
      }
    }
  }

  float* Cb = g.C ? g.C + (long long)(batch >> g.c_bsh) * g.c_bo +
                          (long long)(batch & g.c_bmk) * g.c_bi : nullptr;
  unsigned short* Fb = g.Cbf ? g.Cbf + (long long)(batch >> g.cb_bsh) * g.cb_bo +
                                       (long long)(batch & g.cb_bmk) * g.cb_bi : nullptr;
  const float* Rb = g.R ? g.R + (long long)(batch >> g.r_bsh) * g.r_bo +
                                (long long)(batch & g.r_bmk) * g.r_bi : nullptr;
#pragma unroll
  for (int t = 0; t < 4; ++t) {
    const int col = n0 + t * 16 + mn;
    const float bval = g.bias ? g.bias[col] : 0.0f;
#pragma unroll
    for (int r = 0; r < 8; ++r) {
      const int row = m0 + r + half * 8;
      float v = acc[t][r] + bval;
      if (g.act == 1)      v = fmaxf(v, 0.0f);
      else if (g.act == 2) v = gelu_exact(v);
      if (Rb) v += Rb[(long long)row * g.r_rs + (long long)col * g.r_cs];
      if (Cb) Cb[(long long)row * g.c_rs + (long long)col * g.c_cs] = v;
      if (Fb) Fb[(long long)row * g.cb_rs + (long long)col * g.cb_cs] = f2bf(v * g.bf_scale);
    }
  }
}

// ---------------------------------------------------------------------------
// Fused CPB relative-position MLP: feat(2) -> relu 128 -> relu 128 -> 1
// One 16-sample tile per wave; layer-1 (128x128) via WMMA, weights staged in
// LDS *transposed* ([n][k], padded stride 136 for bank rotation) so every B
// fragment is two ds_load_b128 instead of 16 ds_load_u16.
// grid: (16 batches, 512), block 256.
// ---------------------------------------------------------------------------
#define W1_LDS_STRIDE 136

__global__ __launch_bounds__(256)
void cpb_kernel(const float* __restrict__ grid_kv,   // [16][64][2]
                const float* __restrict__ w0, const float* __restrict__ b0,
                const float* __restrict__ w1, const float* __restrict__ b1,
                const float* __restrict__ w2, const float* __restrict__ b2,
                float* __restrict__ bias_out) {      // [16][65536]
  __shared__ unsigned short sW1t[128 * W1_LDS_STRIDE];  // [n][k] transposed
  __shared__ float sw0[256], sb0[128], sb1[128], sw2[128], skv[128];
  const int tid = threadIdx.x;
  const int batch = blockIdx.x;
  for (int idx = tid; idx < 128 * 128; idx += 256) {
    const int n = idx >> 7, k = idx & 127;
    sW1t[n * W1_LDS_STRIDE + k] = f2bf(w1[k * 128 + n]);
  }
  sw0[tid] = w0[tid & 255];
  if (tid < 128) { sb0[tid] = b0[tid]; sb1[tid] = b1[tid]; sw2[tid] = w2[tid]; }
  if (tid >= 128 && tid < 256) skv[tid - 128] = grid_kv[batch * 128 + tid - 128];
  __syncthreads();

  const float b2v = b2[0];
  const int lane = tid & 31, wave = tid >> 5;
  const int half = lane >> 4, mn = lane & 15;
  const int s0 = (blockIdx.y * 8 + wave) * 16;
  const int s = s0 + mn;
  const int i = s >> 6, j = s & 63;

  const float qx = 2.0f * (float)(i & 31) * (1.0f / 31.0f) - 1.0f;
  const float qy = 2.0f * (float)(i >> 5) * (1.0f / 31.0f) - 1.0f;
  const float p0 = qx - skv[j * 2 + 0];
  const float p1 = qy - skv[j * 2 + 1];
  const float f0 = copysignf(log1pf(fabsf(p0)), p0);
  const float f1 = copysignf(log1pf(fabsf(p1)), p1);

  // layer-0 directly into A fragments (each lane owns row m = lane&15)
  v16us afr[4];
#pragma unroll
  for (int st = 0; st < 4; ++st) {
#pragma unroll
    for (int v = 0; v < 8; ++v) {
      const int k = st * 32 + kidx_a(v, half);
      const float ha = fmaxf(f0 * sw0[k]     + f1 * sw0[128 + k]     + sb0[k],     0.0f);
      const float hb = fmaxf(f0 * sw0[k + 1] + f1 * sw0[128 + k + 1] + sb0[k + 1], 0.0f);
      afr[st][2 * v]     = f2bf(ha);
      afr[st][2 * v + 1] = f2bf(hb);
    }
  }

  float partial[8] = {0, 0, 0, 0, 0, 0, 0, 0};
  for (int nt = 0; nt < 8; ++nt) {
    const int n = nt * 16 + mn;
    v8f acc = {};
#pragma unroll
    for (int st = 0; st < 4; ++st) {
      // B fragment: lane n needs K = st*32 + half*16 + {0..15} -> 32B LDS run
      const v8us* bp = (const v8us*)&sW1t[n * W1_LDS_STRIDE + st * 32 + half * 16];
      const v8us blo = bp[0];
      const v8us bhi = bp[1];
      v16us bu;
#pragma unroll
      for (int e = 0; e < 8; ++e) { bu[e] = blo[e]; bu[8 + e] = bhi[e]; }
      acc = wmma_bf16(afr[st], bu, acc);
    }
    const float wv = sw2[n], bb = sb1[n];
#pragma unroll
    for (int r = 0; r < 8; ++r)
      partial[r] += fmaxf(acc[r] + bb, 0.0f) * wv;
  }
#pragma unroll
  for (int r = 0; r < 8; ++r) {
    float pv = partial[r];
    for (int off = 1; off < 16; off <<= 1) pv += __shfl_xor(pv, off, 32);
    partial[r] = pv;
  }
  if (mn == 0) {
#pragma unroll
    for (int r = 0; r < 8; ++r)
      bias_out[(long long)batch * 65536 + s0 + half * 8 + r] = partial[r] + b2v;
  }
}

// ---------------------------------------------------------------------------
// Small helper kernels
// ---------------------------------------------------------------------------
__global__ __launch_bounds__(256)
void bn_stats_kernel(const float* __restrict__ x, float* __restrict__ mean,
                     float* __restrict__ var, long long c_stride,
                     long long o_stride, long long i_stride, int inner_sh, int total) {
  __shared__ float ss[256], ss2[256];
  const int c = blockIdx.x, tid = threadIdx.x;
  float s = 0.0f, s2 = 0.0f;
  for (int t = tid; t < total; t += 256) {
    const int o = t >> inner_sh, ii = t & ((1 << inner_sh) - 1);
    const float v = x[(long long)c * c_stride + (long long)o * o_stride +
                      (long long)ii * i_stride];
    s += v; s2 += v * v;
  }
  ss[tid] = s; ss2[tid] = s2; __syncthreads();
  for (int st = 128; st > 0; st >>= 1) {
    if (tid < st) { ss[tid] += ss[tid + st]; ss2[tid] += ss2[tid + st]; }
    __syncthreads();
  }
  if (tid == 0) {
    const float m = ss[0] / (float)total;
    mean[c] = m;
    var[c] = ss2[0] / (float)total - m * m;
  }
}

__global__ __launch_bounds__(256)
void bn1_apply_kernel(const float* __restrict__ x, const float* __restrict__ mean,
                      const float* __restrict__ var, const float* __restrict__ gg,
                      const float* __restrict__ bb, float* __restrict__ xn,
                      unsigned short* __restrict__ xn_bf) {
  const long long idx = (long long)blockIdx.x * 256 + threadIdx.x;  // [b][c][hw]
  const int c = (int)((idx >> 10) & 511);
  const float v = (x[idx] - mean[c]) * rsqrtf(var[c] + 1e-5f) * gg[c] + bb[c];
  xn[idx] = v;
  xn_bf[idx] = f2bf(v);
}

__global__ __launch_bounds__(256)
void bn2_apply_kernel(const float* __restrict__ x, const float* __restrict__ mean,
                      const float* __restrict__ var, const float* __restrict__ gg,
                      const float* __restrict__ bb, unsigned short* __restrict__ h_bf) {
  const long long idx = (long long)blockIdx.x * 256 + threadIdx.x;  // [t][c]
  const int c = (int)(idx & 511);
  const float v = (x[idx] - mean[c]) * rsqrtf(var[c] + 1e-5f) * gg[c] + bb[c];
  h_bf[idx] = f2bf(v);
}

__global__ __launch_bounds__(256)
void cvt_bf_kernel(const float* __restrict__ src, unsigned short* __restrict__ dst, int n) {
  const int idx = blockIdx.x * 256 + threadIdx.x;
  if (idx < n) dst[idx] = f2bf(src[idx]);
}

// Depthwise 6x6 stride-4 pad-1 conv + bias + exact GELU. out: [16][64][8][8]
__global__ __launch_bounds__(256)
void dwconv_kernel(const float* __restrict__ q, const float* __restrict__ w,
                   const float* __restrict__ bias, float* __restrict__ out) {
  const int idx = blockIdx.x * 256 + threadIdx.x;  // 65536
  const int x = idx & 7, y = (idx >> 3) & 7, c = (idx >> 6) & 63, n = idx >> 12;
  const float* in = q + (long long)(n * 64 + c) * 1024;
  const float* wc = w + c * 36;
  float s = 0.0f;
#pragma unroll
  for (int ky = 0; ky < 6; ++ky) {
    const int iy = y * 4 - 1 + ky;
    if ((unsigned)iy >= 32u) continue;
#pragma unroll
    for (int kx = 0; kx < 6; ++kx) {
      const int ix = x * 4 - 1 + kx;
      if ((unsigned)ix >= 32u) continue;
      s += in[iy * 32 + ix] * wc[ky * 6 + kx];
    }
  }
  s += bias[c];
  out[idx] = gelu_exact(s);
}

// Pointwise 64->2 + tanh*4, then build normalized sample grid [16][64][2]
__global__ __launch_bounds__(256)
void pwgrid_kernel(const float* __restrict__ off1, const float* __restrict__ pw,
                   float* __restrict__ grid_kv) {
  const int idx = blockIdx.x * 256 + threadIdx.x;  // 1024
  if (idx >= 1024) return;
  const int x = idx & 7, y = (idx >> 3) & 7, n = idx >> 6;
  float s0 = 0.0f, s1 = 0.0f;
  for (int c = 0; c < 64; ++c) {
    const float v = off1[((n * 64 + c) * 8 + y) * 8 + x];
    s0 += v * pw[c];
    s1 += v * pw[64 + c];
  }
  const float ox = tanhf(s0) * 4.0f;
  const float oy = tanhf(s1) * 4.0f;
  const int j = y * 8 + x;
  grid_kv[(n * 64 + j) * 2 + 0] = 2.0f * ((float)x + ox) * (1.0f / 7.0f) - 1.0f;
  grid_kv[(n * 64 + j) * 2 + 1] = 2.0f * ((float)y + oy) * (1.0f / 7.0f) - 1.0f;
}

// Bilinear grid-sample of xn (zero padded) -> kv_bf [16][64][64] bf16
__global__ __launch_bounds__(256)
void gridsample_kernel(const float* __restrict__ xn, const float* __restrict__ grid_kv,
                       unsigned short* __restrict__ kv_bf) {
  const int idx = blockIdx.x * 256 + threadIdx.x;  // 65536
  const int j = idx & 63, c = (idx >> 6) & 63, n = idx >> 12;
  const float vx = grid_kv[(n * 64 + j) * 2 + 0];
  const float vy = grid_kv[(n * 64 + j) * 2 + 1];
  const float xs = ((vx + 1.0f) * 32.0f - 1.0f) * 0.5f;
  const float ys = ((vy + 1.0f) * 32.0f - 1.0f) * 0.5f;
  const float x0f = floorf(xs), y0f = floorf(ys);
  const float wx = xs - x0f, wy = ys - y0f;
  const int x0 = (int)x0f, y0 = (int)y0f;
  const float* im = xn + (long long)(n * 64 + c) * 1024;
  auto tap = [&](int xx, int yy) -> float {
    return ((unsigned)xx < 32u && (unsigned)yy < 32u) ? im[yy * 32 + xx] : 0.0f;
  };
  const float v00 = tap(x0, y0), v10 = tap(x0 + 1, y0);
  const float v01 = tap(x0, y0 + 1), v11 = tap(x0 + 1, y0 + 1);
  const float r = v00 * (1 - wx) * (1 - wy) + v10 * wx * (1 - wy) +
                  v01 * (1 - wx) * wy + v11 * wx * wy;
  kv_bf[(long long)(n * 64 + c) * 64 + j] = f2bf(r);
}

// Row softmax over j=64 of (sim + cpb_bias); one wave per row -> bf16 attn
__global__ __launch_bounds__(256)
void softmax_kernel(const float* __restrict__ sim, const float* __restrict__ cbias,
                    unsigned short* __restrict__ attn_bf) {
  const int wave = threadIdx.x >> 5, lane = threadIdx.x & 31;
  const long long row = (long long)blockIdx.x * 8 + wave;  // 16384 rows
  const float* s = sim + row * 64;
  const float* b = cbias + row * 64;
  float v0 = s[lane] + b[lane];
  float v1 = s[lane + 32] + b[lane + 32];
  float m = fmaxf(v0, v1);
  for (int off = 16; off >= 1; off >>= 1) m = fmaxf(m, __shfl_xor(m, off, 32));
  const float e0 = __expf(v0 - m), e1 = __expf(v1 - m);
  float sum = e0 + e1;
  for (int off = 16; off >= 1; off >>= 1) sum += __shfl_xor(sum, off, 32);
  const float inv = 1.0f / sum;
  attn_bf[row * 64 + lane]      = f2bf(e0 * inv);
  attn_bf[row * 64 + lane + 32] = f2bf(e1 * inv);
}

// ---------------------------------------------------------------------------
// Host launch
// ---------------------------------------------------------------------------
static void launch_gemm(hipStream_t stream, GArgs g, int batches,
                        int M, int N, int K,
                        int a_bic, int b_bic, int c_bic, int cb_bic, int r_bic) {
  const int n_super = N / 64;
  g.K = K;
  g.ns_shift = __builtin_ctz(n_super);
  g.ns_mask = n_super - 1;
  g.tiles_per_batch = (M / 16) * n_super;
  g.a_bsh = __builtin_ctz(a_bic);  g.a_bmk = a_bic - 1;
  g.b_bsh = __builtin_ctz(b_bic);  g.b_bmk = b_bic - 1;
  g.c_bsh = __builtin_ctz(c_bic);  g.c_bmk = c_bic - 1;
  g.cb_bsh = __builtin_ctz(cb_bic); g.cb_bmk = cb_bic - 1;
  g.r_bsh = __builtin_ctz(r_bic);  g.r_bmk = r_bic - 1;
  dim3 grid(batches, (g.tiles_per_batch + 7) / 8);
  wmma_gemm_kernel<<<grid, dim3(256), 0, stream>>>(g);
}

extern "C" void kernel_launch(void* const* d_in, const int* in_sizes, int n_in,
                              void* d_out, int out_size, void* d_ws, size_t ws_size,
                              hipStream_t stream) {
  (void)in_sizes; (void)n_in; (void)out_size; (void)ws_size;
  const float* x        = (const float*)d_in[0];
  const float* n1_g     = (const float*)d_in[1];
  const float* n1_b     = (const float*)d_in[2];
  const float* n2_g     = (const float*)d_in[3];
  const float* n2_b     = (const float*)d_in[4];
  const float* q_w      = (const float*)d_in[5];
  const float* k_w      = (const float*)d_in[6];
  const float* v_w      = (const float*)d_in[7];
  const float* off_dw_w = (const float*)d_in[8];
  const float* off_dw_b = (const float*)d_in[9];
  const float* off_pw_w = (const float*)d_in[10];
  const float* cpb_w0   = (const float*)d_in[11];
  const float* cpb_b0   = (const float*)d_in[12];
  const float* cpb_w1   = (const float*)d_in[13];
  const float* cpb_b1   = (const float*)d_in[14];
  const float* cpb_w2   = (const float*)d_in[15];
  const float* cpb_b2   = (const float*)d_in[16];
  const float* out_w    = (const float*)d_in[17];
  const float* out_b    = (const float*)d_in[18];
  const float* mlp_w1   = (const float*)d_in[19];
  const float* mlp_b1   = (const float*)d_in[20];
  const float* mlp_w2   = (const float*)d_in[21];
  const float* mlp_b2   = (const float*)d_in[22];
  float* out = (float*)d_out;

  // Workspace bump allocator (256B aligned)
  char* W = (char*)d_ws;
  size_t off = 0;
  auto alloc = [&](size_t bytes) -> void* {
    void* p = (void*)(W + off);
    off += (bytes + 255) & ~(size_t)255;
    return p;
  };
  float* m1      = (float*)alloc(512 * 4);
  float* v1      = (float*)alloc(512 * 4);
  float* m2      = (float*)alloc(512 * 4);
  float* v2      = (float*)alloc(512 * 4);
  float* xn      = (float*)alloc(1048576ull * 4);          // [b][c][hw] fp32
  unsigned short* xn_bf  = (unsigned short*)alloc(1048576ull * 2);
  float* qbuf    = (float*)alloc(1048576ull * 4);          // [b][g][o][hw]
  unsigned short* qs_bf  = (unsigned short*)alloc(1048576ull * 2); // [bh][i][d]*SCALE
  float* off1    = (float*)alloc(65536ull * 4);            // [16][64][8][8]
  float* gkv     = (float*)alloc(2048ull * 4);             // [16][64][2]
  unsigned short* kv_bf  = (unsigned short*)alloc(65536ull * 2);   // [16][64][64]
  unsigned short* k_bf   = (unsigned short*)alloc(65536ull * 2);   // [bh][d][j]
  unsigned short* v_bf   = (unsigned short*)alloc(65536ull * 2);   // [bh][d][j]
  float* sim     = (float*)alloc(1048576ull * 4);          // [bh][i][j]
  float* cbias   = (float*)alloc(1048576ull * 4);          // [bh][i*64+j]
  unsigned short* attn_bf = (unsigned short*)alloc(1048576ull * 2);
  unsigned short* att_tok = (unsigned short*)alloc(1048576ull * 2); // [b][hw][c]
  float* x2_tok  = (float*)alloc(1048576ull * 4);          // [b][hw][c]
  unsigned short* h_bf   = (unsigned short*)alloc(1048576ull * 2);  // [t][c]
  unsigned short* h1_bf  = (unsigned short*)alloc(4194304ull * 2);  // [t][4c]
  unsigned short* qw_bf  = (unsigned short*)alloc(32768ull * 2);
  unsigned short* kw_bf  = (unsigned short*)alloc(32768ull * 2);
  unsigned short* vw_bf  = (unsigned short*)alloc(32768ull * 2);
  unsigned short* ow_bf  = (unsigned short*)alloc(262144ull * 2);
  unsigned short* mw1_bf = (unsigned short*)alloc(1048576ull * 2);
  unsigned short* mw2_bf = (unsigned short*)alloc(1048576ull * 2);

  // 1. Weight conversions to bf16
  cvt_bf_kernel<<<128,  256, 0, stream>>>(q_w,    qw_bf,  32768);
  cvt_bf_kernel<<<128,  256, 0, stream>>>(k_w,    kw_bf,  32768);
  cvt_bf_kernel<<<128,  256, 0, stream>>>(v_w,    vw_bf,  32768);
  cvt_bf_kernel<<<1024, 256, 0, stream>>>(out_w,  ow_bf,  262144);
  cvt_bf_kernel<<<4096, 256, 0, stream>>>(mlp_w1, mw1_bf, 1048576);
  cvt_bf_kernel<<<4096, 256, 0, stream>>>(mlp_w2, mw2_bf, 1048576);

  // 2. BatchNorm 1 (over b,hw per channel; x layout [b][c][hw])
  bn_stats_kernel<<<512, 256, 0, stream>>>(x, m1, v1, 1024, 524288, 1, 10, 2048);
  bn1_apply_kernel<<<4096, 256, 0, stream>>>(x, m1, v1, n1_g, n1_b, xn, xn_bf);

  // 3. q grouped conv: per (b,g): q[o,hw] = q_w[g] @ xn[b,g]
  {
    GArgs g{};
    g.A = qw_bf; g.a_rs = 64;   g.a_bo = 0;      g.a_bi = 4096;
    g.B = xn_bf; g.b_rs = 1024; g.b_cs = 1; g.b_bo = 524288; g.b_bi = 65536;
    g.C = qbuf;  g.c_rs = 1024; g.c_cs = 1; g.c_bo = 65536;  g.c_bi = 0;
    g.Cbf = qs_bf; g.cb_rs = 1; g.cb_cs = 64; g.cb_bo = 65536; g.cb_bi = 0;
    g.bias = nullptr; g.R = nullptr; g.act = 0; g.bf_scale = 0.125f; // SCALE
    launch_gemm(stream, g, 16, 64, 1024, 64, /*bic*/ 8, 8, 1, 1, 1);
  }

  // 4. Offset path: depthwise conv + gelu, pointwise + tanh, sample grid
  dwconv_kernel<<<256, 256, 0, stream>>>(qbuf, off_dw_w, off_dw_b, off1);
  pwgrid_kernel<<<4, 256, 0, stream>>>(off1, off_pw_w, gkv);

  // 5. Bilinear sampling of normalized x at deformed grid -> kv (bf16)
  gridsample_kernel<<<256, 256, 0, stream>>>(xn, gkv, kv_bf);

  // 6. k and v grouped convs on kv (per (b,g), M=N=K=64)
  {
    GArgs g{};
    g.A = kw_bf; g.a_rs = 64; g.a_bo = 0;    g.a_bi = 4096;
    g.B = kv_bf; g.b_rs = 64; g.b_cs = 1; g.b_bo = 4096; g.b_bi = 0;
    g.C = nullptr;
    g.Cbf = k_bf; g.cb_rs = 64; g.cb_cs = 1; g.cb_bo = 4096; g.cb_bi = 0;
    g.bias = nullptr; g.R = nullptr; g.act = 0; g.bf_scale = 1.0f;
    launch_gemm(stream, g, 16, 64, 64, 64, 8, 1, 1, 1, 1);
    g.A = vw_bf; g.Cbf = v_bf;
    launch_gemm(stream, g, 16, 64, 64, 64, 8, 1, 1, 1, 1);
  }

  // 7. sim = (q*SCALE)^T k : per (b,h), M=1024(i), N=64(j), K=64(d)
  {
    GArgs g{};
    g.A = qs_bf; g.a_rs = 64; g.a_bo = 65536; g.a_bi = 0;
    g.B = k_bf;  g.b_rs = 64; g.b_cs = 1; g.b_bo = 4096;  g.b_bi = 0;
    g.C = sim;   g.c_rs = 64; g.c_cs = 1; g.c_bo = 65536; g.c_bi = 0;
    g.Cbf = nullptr;
    g.bias = nullptr; g.R = nullptr; g.act = 0; g.bf_scale = 1.0f;
    launch_gemm(stream, g, 16, 1024, 64, 64, 1, 1, 1, 1, 1);
  }

  // 8. Fused CPB positional-bias MLP (dominant FLOPs, WMMA on 128x128 layer)
  cpb_kernel<<<dim3(16, 512), 256, 0, stream>>>(gkv, cpb_w0, cpb_b0, cpb_w1,
                                                cpb_b1, cpb_w2, cpb_b2, cbias);

  // 9. softmax(sim + bias) over j
  softmax_kernel<<<2048, 256, 0, stream>>>(sim, cbias, attn_bf);

  // 10. out = attn @ v^T -> token-major bf16 [b][hw][h*64+d]  (b_rs==1 fast path)
  {
    GArgs g{};
    g.A = attn_bf; g.a_rs = 64; g.a_bo = 65536; g.a_bi = 0;
    g.B = v_bf;    g.b_rs = 1;  g.b_cs = 64; g.b_bo = 4096;  g.b_bi = 0;
    g.C = nullptr;
    g.Cbf = att_tok; g.cb_rs = 512; g.cb_cs = 1; g.cb_bo = 524288; g.cb_bi = 64;
    g.bias = nullptr; g.R = nullptr; g.act = 0; g.bf_scale = 1.0f;
    launch_gemm(stream, g, 16, 1024, 64, 64, 1, 1, 1, /*cb bic*/ 8, 1);
  }

  // 11. Output projection + out_b + residual x -> x2_tok (b_rs==1 fast path)
  {
    GArgs g{};
    g.A = att_tok; g.a_rs = 512; g.a_bo = 524288; g.a_bi = 0;
    g.B = ow_bf;   g.b_rs = 1;   g.b_cs = 512; g.b_bo = 0; g.b_bi = 0;
    g.C = x2_tok;  g.c_rs = 512; g.c_cs = 1;   g.c_bo = 524288; g.c_bi = 0;
    g.Cbf = nullptr;
    g.bias = out_b;
    g.R = x; g.r_rs = 1; g.r_cs = 1024; g.r_bo = 524288; g.r_bi = 0;
    g.act = 0; g.bf_scale = 1.0f;
    launch_gemm(stream, g, 2, 1024, 512, 512, 1, 1, 1, 1, 1);
  }

  // 12. BatchNorm 2 on x2_tok (reduce over tokens per channel)
  bn_stats_kernel<<<512, 256, 0, stream>>>(x2_tok, m2, v2, 1, 0, 512, 11, 2048);
  bn2_apply_kernel<<<4096, 256, 0, stream>>>(x2_tok, m2, v2, n2_g, n2_b, h_bf);

  // 13. MLP up-projection with exact GELU (b_rs==1 fast path)
  {
    GArgs g{};
    g.A = h_bf;   g.a_rs = 512;  g.a_bo = 0; g.a_bi = 0;
    g.B = mw1_bf; g.b_rs = 1;    g.b_cs = 512; g.b_bo = 0; g.b_bi = 0;
    g.C = nullptr;
    g.Cbf = h1_bf; g.cb_rs = 2048; g.cb_cs = 1; g.cb_bo = 0; g.cb_bi = 0;
    g.bias = mlp_b1; g.R = nullptr; g.act = 2; g.bf_scale = 1.0f;
    launch_gemm(stream, g, 1, 2048, 2048, 512, 1, 1, 1, 1, 1);
  }

  // 14. MLP down-projection + mlp_b2 + residual x2 -> d_out (b_rs==1 fast path)
  {
    GArgs g{};
    g.A = h1_bf;  g.a_rs = 2048; g.a_bo = 2097152; g.a_bi = 0;
    g.B = mw2_bf; g.b_rs = 1;    g.b_cs = 2048; g.b_bo = 0; g.b_bi = 0;
    g.C = out;    g.c_rs = 1;    g.c_cs = 1024; g.c_bo = 524288; g.c_bi = 0;
    g.Cbf = nullptr;
    g.bias = mlp_b2;
    g.R = x2_tok; g.r_rs = 512; g.r_cs = 1; g.r_bo = 524288; g.r_bi = 0;
    g.act = 0; g.bf_scale = 1.0f;
    launch_gemm(stream, g, 2, 1024, 512, 2048, 1, 1, 1, 1, 1);
  }
}
